// PGNNNetSparse_89232240542464
// MI455X (gfx1250) — matrix operand order repaired and checked by reading, not compiled, not run on hardware
//
#include <hip/hip_runtime.h>
#include <hip/hip_bf16.h>

// ---------------------------------------------------------------------------
// PGNNNetSparse for gfx1250 (MI455X).
// Dense GEMMs: v_wmma_f32_16x16x32_f16, 4 waves/block, one 16x16 tile per wave.
//   - A read as uniform float4s from stride-64 zero-padded f32 buffers (no
//     bounds checks in the K loop -> clean global_load_b128 + v_cvt).
//   - W pre-packed to the exact B-matrix per-lane f16 layout (contiguous
//     32B per lane per k-step).
// Sparse propagation: per-(edge,quad) float4 gather + atomicAdd scatter.
// ---------------------------------------------------------------------------

typedef __attribute__((ext_vector_type(16))) _Float16 v16h;
typedef __attribute__((ext_vector_type(8)))  float    v8f;

union V16H { v16h v; _Float16 e[16]; float4 q[2]; };
union V8F  { v8f  v; float    e[8];  };

#define FLAG_RELU  1
#define FLAG_ACCUM 2
#define FLAG_BIAS  4

#define LD 64   // row stride (floats) of all node-feature buffers

// ---- weight pack: W[K,Dout] f32 -> Wp f16 in B-matrix lane layout ----------
// Wp[((nt*ktiles + kt)*32 + lane)*16 + e] = f16(W[(kt*32 + (e<8 ? 8g+e : 16+8g+(e-8)))*Dout + nt*16 + l])
__global__ void pack_w_f16(const float* __restrict__ W, _Float16* __restrict__ Wp,
                           int K, int Kp, int Dout)
{
    int idx = blockIdx.x * blockDim.x + threadIdx.x;
    int ktiles = Kp >> 5;
    int total = (Dout >> 4) * ktiles * 32 * 16;
    if (idx >= total) return;
    int e    = idx & 15;
    int lane = (idx >> 4) & 31;
    int t    = idx >> 9;            // nt*ktiles + kt
    int nt   = t / ktiles;
    int kt   = t % ktiles;
    int g = lane >> 4, l = lane & 15;
    int k = kt * 32 + ((e < 8) ? (8 * g + e) : (16 + 8 * g + (e - 8)));
    Wp[idx] = (k < K) ? (_Float16)W[(size_t)k * Dout + nt * 16 + l] : (_Float16)0.f;
}

// ---- WMMA GEMM: C[M,Dout] (op)= act(A[M,Kp] @ W + bias) --------------------
// A: f32, row stride lda, columns >= real K are zero. Kp % 32 == 0.
// 4 waves per block, each wave one 16x16 tile.
__global__ __launch_bounds__(128) void gemm_wmma(
    const float* __restrict__ A, int lda,
    const _Float16* __restrict__ Wp,
    const float* __restrict__ bias,
    float* __restrict__ C, int ldc,
    int Mtiles, int Kp, int Dout, int flags)
{
    const int ntiles = Dout >> 4;
    const int ktiles = Kp >> 5;
    const int wave = threadIdx.x >> 5;
    const int tile = blockIdx.x * 4 + wave;
    if (tile >= Mtiles * ntiles) return;      // wave-uniform guard
    const int mt = tile / ntiles;
    const int nt = tile % ntiles;
    const int lane = threadIdx.x & 31;
    const int g = lane >> 4;
    const int l = lane & 15;
    const int m0 = mt << 4;
    const int n0 = nt << 4;

    V8F c;
    if (flags & FLAG_ACCUM) {
#pragma unroll
        for (int r = 0; r < 8; ++r)
            c.e[r] = C[(size_t)(m0 + r + 8 * g) * ldc + n0 + l];
    } else {
#pragma unroll
        for (int r = 0; r < 8; ++r) c.e[r] = 0.f;
    }

    const float* arow = A + (size_t)(m0 + l) * lda;
    const _Float16* bp = Wp + ((size_t)nt * ktiles * 32 + lane) * 16;

    for (int kt = 0; kt < ktiles; ++kt) {
        const int kb = kt * 32;
        // A 16x32 f16 layout: e<8 -> K=kb+8g+e ; e>=8 -> K=kb+16+8g+(e-8)
        float4 a0 = *(const float4*)(arow + kb + 8 * g);
        float4 a1 = *(const float4*)(arow + kb + 8 * g + 4);
        float4 a2 = *(const float4*)(arow + kb + 16 + 8 * g);
        float4 a3 = *(const float4*)(arow + kb + 16 + 8 * g + 4);
        V16H a;
        a.e[0]  = (_Float16)a0.x; a.e[1]  = (_Float16)a0.y;
        a.e[2]  = (_Float16)a0.z; a.e[3]  = (_Float16)a0.w;
        a.e[4]  = (_Float16)a1.x; a.e[5]  = (_Float16)a1.y;
        a.e[6]  = (_Float16)a1.z; a.e[7]  = (_Float16)a1.w;
        a.e[8]  = (_Float16)a2.x; a.e[9]  = (_Float16)a2.y;
        a.e[10] = (_Float16)a2.z; a.e[11] = (_Float16)a2.w;
        a.e[12] = (_Float16)a3.x; a.e[13] = (_Float16)a3.y;
        a.e[14] = (_Float16)a3.z; a.e[15] = (_Float16)a3.w;
        // B: contiguous 16 halves per lane (pre-packed layout)
        V16H b;
        const float4* bq = (const float4*)(bp + (size_t)(kt * 32) * 16);
        b.q[0] = bq[0];
        b.q[1] = bq[1];
        c.v = __builtin_amdgcn_wmma_f32_16x16x32_f16(
            false, a.v, false, b.v, (short)0, c.v, false, false);
    }

#pragma unroll
    for (int r = 0; r < 8; ++r) {
        float v = c.e[r];
        if (flags & FLAG_BIAS) v += bias[n0 + l];
        if (flags & FLAG_RELU) v = v > 0.f ? v : 0.f;
        C[(size_t)(m0 + r + 8 * g) * ldc + n0 + l] = v;
    }
}

// ---- sparse-graph helpers --------------------------------------------------
__global__ void deg_kernel(const int* __restrict__ row, const int* __restrict__ col,
                           const float* __restrict__ w,
                           float* __restrict__ deg_c, float* __restrict__ deg_r, int E)
{
    int e = blockIdx.x * blockDim.x + threadIdx.x;
    if (e >= E) return;
    float we = w[e];
    atomicAdd(&deg_c[col[e]], we);
    atomicAdd(&deg_r[row[e]], we);
}

__global__ void dinv_kernel(const float* __restrict__ deg_c, const float* __restrict__ deg_r,
                            float* __restrict__ dinv_c, float* __restrict__ dinv_r, int N)
{
    int i = blockIdx.x * blockDim.x + threadIdx.x;
    if (i >= N) return;
    float dc = deg_c[i], dr = deg_r[i];
    dinv_c[i] = dc > 0.f ? rsqrtf(dc) : 0.f;
    dinv_r[i] = dr > 0.f ? rsqrtf(dr) : 0.f;
}

// cg = dinv_c[row]*w*dinv_c[col] ; lw = -(dinv_r[row]*w*dinv_r[col]) - (row==col)
__global__ void coef_kernel(const int* __restrict__ row, const int* __restrict__ col,
                            const float* __restrict__ w,
                            const float* __restrict__ dinv_c, const float* __restrict__ dinv_r,
                            float* __restrict__ cg, float* __restrict__ lw, int E)
{
    int e = blockIdx.x * blockDim.x + threadIdx.x;
    if (e >= E) return;
    int r = row[e], c = col[e];
    float we = w[e];
    cg[e] = dinv_c[r] * we * dinv_c[c];
    float nw = dinv_r[r] * we * dinv_r[c];
    lw[e] = -nw - ((r == c) ? 1.f : 0.f);
}

// dst[col[e], :48] += scale*coef[e]*src[row[e], :48]; thread = (edge, quad of 4)
__global__ void scatter_edges(const int* __restrict__ row, const int* __restrict__ col,
                              const float* __restrict__ coef, float scale,
                              const float* __restrict__ src, float* __restrict__ dst, int E)
{
    int idx = blockIdx.x * blockDim.x + threadIdx.x;
    int e = idx / 12;
    int q = idx % 12;
    if (e >= E) return;
    int r = row[e], c = col[e];
    float cf = coef[e] * scale;
    float4 v = *(const float4*)(src + (size_t)r * LD + q * 4);
    float* d = dst + (size_t)c * LD + q * 4;
    atomicAdd(d + 0, cf * v.x);
    atomicAdd(d + 1, cf * v.y);
    atomicAdd(d + 2, cf * v.z);
    atomicAdd(d + 3, cf * v.w);
}

// buf[i, f<48] = act(buf[i,f] + bias[f])   (stride LD)
__global__ void bias_act_kernel(float* __restrict__ buf, const float* __restrict__ bias,
                                int N, int relu)
{
    int idx = blockIdx.x * blockDim.x + threadIdx.x;
    if (idx >= N * 48) return;
    int i = idx / 48, f = idx % 48;
    float v = buf[(size_t)i * LD + f] + bias[f];
    if (relu) v = v > 0.f ? v : 0.f;
    buf[(size_t)i * LD + f] = v;
}

// a -= b over full padded buffers (pads are 0-0=0)
__global__ void sub_kernel(float* __restrict__ a, const float* __restrict__ b, int n)
{
    int i = blockIdx.x * blockDim.x + threadIdx.x;
    if (i >= n) return;
    a[i] -= b[i];
}

// fcat[i,0:48]=fl[i,:48]; fcat[i,48:96]=fg[i,:48]   (fl/fg stride LD, fcat stride 96)
__global__ void concat_kernel(const float* __restrict__ fl, const float* __restrict__ fg,
                              float* __restrict__ out, int N)
{
    int idx = blockIdx.x * blockDim.x + threadIdx.x;
    if (idx >= N * 96) return;
    int i = idx / 96, f = idx % 96;
    out[idx] = (f < 48) ? fl[(size_t)i * LD + f] : fg[(size_t)i * LD + (f - 48)];
}

__global__ void pool_accum(const float* __restrict__ x, const int* __restrict__ batch,
                           float* __restrict__ sums, float* __restrict__ cnt, int N)
{
    int i = blockIdx.x * blockDim.x + threadIdx.x;
    if (i >= N) return;
    int b = batch[i];
    atomicAdd(&cnt[b], 1.f);
    const float* xr = x + (size_t)i * LD;
    float* sr = sums + (size_t)b * 48;
    for (int f = 0; f < 48; ++f) atomicAdd(&sr[f], xr[f]);
}

__global__ void pool_div(const float* __restrict__ sums, const float* __restrict__ cnt,
                         float* __restrict__ pooled, int G)
{
    int idx = blockIdx.x * blockDim.x + threadIdx.x;
    if (idx >= G * 48) return;
    pooled[idx] = sums[idx] / fmaxf(cnt[idx / 48], 1.f);
}

__global__ void cls_head(const float* __restrict__ pooled,
                         const float* __restrict__ w0, const float* __restrict__ b0,
                         const float* __restrict__ w1, const float* __restrict__ b1,
                         float* __restrict__ out, int G, int D, int NCLS)
{
    int idx = blockIdx.x * blockDim.x + threadIdx.x;
    if (idx >= G * NCLS) return;
    int g = idx / NCLS, cl = idx % NCLS;
    float acc = b1[cl];
    for (int j = 0; j < D; ++j) {
        float h = b0[j];
        for (int k = 0; k < D; ++k) h += pooled[(size_t)g * D + k] * w0[(size_t)k * D + j];
        h = fmaxf(h, 0.f);
        acc += h * w1[(size_t)j * NCLS + cl];
    }
    out[idx] = acc;
}

static inline int cdiv(int a, int b) { return (a + b - 1) / b; }

extern "C" void kernel_launch(void* const* d_in, const int* in_sizes, int n_in,
                              void* d_out, int out_size, void* d_ws, size_t ws_size,
                              hipStream_t stream)
{
    const float* x1   = (const float*)d_in[0];
    const float* x2   = (const float*)d_in[1];
    const int*   eidx = (const int*)  d_in[2];
    const int*   batch= (const int*)  d_in[3];
    const float* ew   = (const float*)d_in[4];
    const float* enc1_w0 = (const float*)d_in[5];
    const float* enc1_b0 = (const float*)d_in[6];
    const float* enc1_w1 = (const float*)d_in[7];
    const float* enc1_b1 = (const float*)d_in[8];
    const float* enc2_w0 = (const float*)d_in[9];
    const float* enc2_b0 = (const float*)d_in[10];
    const float* enc2_w1 = (const float*)d_in[11];
    const float* enc2_b1 = (const float*)d_in[12];
    const float* gcn1_w  = (const float*)d_in[13];
    const float* gcn1_b  = (const float*)d_in[14];
    const float* gcn2_w  = (const float*)d_in[15];
    const float* gcn2_b  = (const float*)d_in[16];
    const float* cheb1_w = (const float*)d_in[17];  // [3,48,48]
    const float* cheb1_b = (const float*)d_in[18];
    const float* cheb2_w = (const float*)d_in[19];  // [3,48,48]
    const float* cheb2_b = (const float*)d_in[20];
    const float* fuse_w  = (const float*)d_in[21];  // [96,48]
    const float* fuse_b  = (const float*)d_in[22];
    const float* cls_w0  = (const float*)d_in[23];
    const float* cls_b0  = (const float*)d_in[24];
    const float* cls_w1  = (const float*)d_in[25];
    const float* cls_b1  = (const float*)d_in[26];
    float* logits = (float*)d_out;

    const int N = in_sizes[3];
    const int E = in_sizes[4];
    const int G = 64, NCLS = 10;
    const int* row = eidx;
    const int* col = eidx + E;

    // ---- workspace carve ----
    float* ws = (float*)d_ws;
    size_t off = 0;
    auto alloc = [&](size_t nf) { float* p = ws + off; off += (nf + 15) & ~(size_t)15; return p; };
    float* tmp   = alloc((size_t)N * LD);    // also encoder hidden [N,64]
    float* h1    = alloc((size_t)N * LD);
    float* h2    = alloc((size_t)N * LD);
    float* gbuf  = alloc((size_t)N * LD);
    float* floc  = alloc((size_t)N * LD);
    float* cbuf  = alloc((size_t)N * LD);
    float* fglob = alloc((size_t)N * LD);
    float* t1    = alloc((size_t)N * LD);
    float* t2    = alloc((size_t)N * LD);
    float* fused = alloc((size_t)N * LD);
    float* fcat  = alloc((size_t)N * 96);
    float* deg_c = alloc(N);
    float* deg_r = alloc(N);
    float* dinv_c= alloc(N);
    float* dinv_r= alloc(N);
    float* cg    = alloc(E);
    float* lw    = alloc(E);
    float* psum  = alloc((size_t)G * 48);
    float* pcnt  = alloc(G);
    float* pooled= alloc((size_t)G * 48);
    // packed f16 weights (sizes in halves; alloc() counts floats -> /2, keep 64B align)
    auto allocH = [&](size_t nh) { return (_Float16*)alloc((nh + 1) / 2); };
    const size_t PS_64_64 = 4 * 2 * 512;   // K=64,Dout=64
    const size_t PS_64_48 = 3 * 2 * 512;   // K=64(p),Dout=48
    const size_t PS_96_48 = 3 * 3 * 512;   // K=96,Dout=48
    _Float16* p_e1w0 = allocH(PS_64_64);
    _Float16* p_e1w1 = allocH(PS_64_48);
    _Float16* p_e2w0 = allocH(PS_64_64);
    _Float16* p_e2w1 = allocH(PS_64_48);
    _Float16* p_g1   = allocH(PS_64_48);
    _Float16* p_g2   = allocH(PS_64_48);
    _Float16* p_c1[3], *p_c2[3];
    for (int k = 0; k < 3; ++k) { p_c1[k] = allocH(PS_64_48); }
    for (int k = 0; k < 3; ++k) { p_c2[k] = allocH(PS_64_48); }
    _Float16* p_fu   = allocH(PS_96_48);
    (void)ws_size; (void)n_in; (void)out_size;

    const int TB = 256;
    const int ebl  = cdiv(E, TB);
    const int nbl  = cdiv(N, TB);
    const int sbl  = cdiv(E * 12, TB);
    const size_t NLD_bytes = (size_t)N * LD * sizeof(float);
    const int Mtiles = N / 16;
    auto ggrid = [&](int Dout) { return cdiv(Mtiles * (Dout / 16), 4); };

    // ---- zero pads / accumulators ----
    hipMemsetAsync(tmp,  0, NLD_bytes, stream);
    hipMemsetAsync(h1,   0, NLD_bytes, stream);
    hipMemsetAsync(h2,   0, NLD_bytes, stream);
    hipMemsetAsync(cbuf, 0, NLD_bytes, stream);
    hipMemsetAsync(fglob,0, NLD_bytes, stream);
    hipMemsetAsync(fused,0, NLD_bytes, stream);
    hipMemsetAsync(deg_c,0, (size_t)N * sizeof(float), stream);
    hipMemsetAsync(deg_r,0, (size_t)N * sizeof(float), stream);
    hipMemsetAsync(psum, 0, (size_t)G * 48 * sizeof(float), stream);
    hipMemsetAsync(pcnt, 0, (size_t)G * sizeof(float), stream);

    // ---- pack weights ----
    pack_w_f16<<<cdiv((int)PS_64_64, TB), TB, 0, stream>>>(enc1_w0, p_e1w0, 64, 64, 64);
    pack_w_f16<<<cdiv((int)PS_64_48, TB), TB, 0, stream>>>(enc1_w1, p_e1w1, 64, 64, 48);
    pack_w_f16<<<cdiv((int)PS_64_64, TB), TB, 0, stream>>>(enc2_w0, p_e2w0, 64, 64, 64);
    pack_w_f16<<<cdiv((int)PS_64_48, TB), TB, 0, stream>>>(enc2_w1, p_e2w1, 64, 64, 48);
    pack_w_f16<<<cdiv((int)PS_64_48, TB), TB, 0, stream>>>(gcn1_w, p_g1, 48, 64, 48);
    pack_w_f16<<<cdiv((int)PS_64_48, TB), TB, 0, stream>>>(gcn2_w, p_g2, 48, 64, 48);
    for (int k = 0; k < 3; ++k)
        pack_w_f16<<<cdiv((int)PS_64_48, TB), TB, 0, stream>>>(cheb1_w + k * 48 * 48, p_c1[k], 48, 64, 48);
    for (int k = 0; k < 3; ++k)
        pack_w_f16<<<cdiv((int)PS_64_48, TB), TB, 0, stream>>>(cheb2_w + k * 48 * 48, p_c2[k], 48, 64, 48);
    pack_w_f16<<<cdiv((int)PS_96_48, TB), TB, 0, stream>>>(fuse_w, p_fu, 96, 96, 48);

    // ---- edge coefficients ----
    deg_kernel<<<ebl, TB, 0, stream>>>(row, col, ew, deg_c, deg_r, E);
    dinv_kernel<<<nbl, TB, 0, stream>>>(deg_c, deg_r, dinv_c, dinv_r, N);
    coef_kernel<<<ebl, TB, 0, stream>>>(row, col, ew, dinv_c, dinv_r, cg, lw, E);

    // ---- encoders ----
    gemm_wmma<<<ggrid(64), 128, 0, stream>>>(x1, 64, p_e1w0, enc1_b0, tmp, LD, Mtiles, 64, 64, FLAG_BIAS | FLAG_RELU);
    gemm_wmma<<<ggrid(48), 128, 0, stream>>>(tmp, LD, p_e1w1, enc1_b1, h1, LD, Mtiles, 64, 48, FLAG_BIAS);
    gemm_wmma<<<ggrid(64), 128, 0, stream>>>(x2, 64, p_e2w0, enc2_b0, tmp, LD, Mtiles, 64, 64, FLAG_BIAS | FLAG_RELU);
    gemm_wmma<<<ggrid(48), 128, 0, stream>>>(tmp, LD, p_e2w1, enc2_b1, h2, LD, Mtiles, 64, 48, FLAG_BIAS);

    // ---- GCN branch ----
    hipMemsetAsync(tmp, 0, NLD_bytes, stream);  // re-zero pad (cols 48..63 were written above)
    gemm_wmma<<<ggrid(48), 128, 0, stream>>>(h1, LD, p_g1, nullptr, tmp, LD, Mtiles, 64, 48, 0);
    hipMemsetAsync(gbuf, 0, NLD_bytes, stream);
    scatter_edges<<<sbl, TB, 0, stream>>>(row, col, cg, 1.f, tmp, gbuf, E);
    bias_act_kernel<<<cdiv(N * 48, TB), TB, 0, stream>>>(gbuf, gcn1_b, N, 1);

    gemm_wmma<<<ggrid(48), 128, 0, stream>>>(gbuf, LD, p_g2, nullptr, tmp, LD, Mtiles, 64, 48, 0);
    hipMemsetAsync(floc, 0, NLD_bytes, stream);
    scatter_edges<<<sbl, TB, 0, stream>>>(row, col, cg, 1.f, tmp, floc, E);
    bias_act_kernel<<<cdiv(N * 48, TB), TB, 0, stream>>>(floc, gcn2_b, N, 1);

    // ---- Cheb branch (K=3): out = x@W0 + Lhat(x)@W1 + (2*Lhat(Lhat(x)) - x)@W2 + b
    // cheb1: x = h2 -> cbuf
    gemm_wmma<<<ggrid(48), 128, 0, stream>>>(h2, LD, p_c1[0], nullptr, cbuf, LD, Mtiles, 64, 48, 0);
    hipMemsetAsync(t1, 0, NLD_bytes, stream);
    scatter_edges<<<sbl, TB, 0, stream>>>(row, col, lw, 1.f, h2, t1, E);
    gemm_wmma<<<ggrid(48), 128, 0, stream>>>(t1, LD, p_c1[1], nullptr, cbuf, LD, Mtiles, 64, 48, FLAG_ACCUM);
    hipMemsetAsync(t2, 0, NLD_bytes, stream);
    scatter_edges<<<sbl, TB, 0, stream>>>(row, col, lw, 2.f, t1, t2, E);
    sub_kernel<<<cdiv(N * LD, TB), TB, 0, stream>>>(t2, h2, N * LD);
    gemm_wmma<<<ggrid(48), 128, 0, stream>>>(t2, LD, p_c1[2], cheb1_b, cbuf, LD, Mtiles, 64, 48,
                                             FLAG_ACCUM | FLAG_BIAS | FLAG_RELU);
    // cheb2: x = cbuf -> fglob
    gemm_wmma<<<ggrid(48), 128, 0, stream>>>(cbuf, LD, p_c2[0], nullptr, fglob, LD, Mtiles, 64, 48, 0);
    hipMemsetAsync(t1, 0, NLD_bytes, stream);
    scatter_edges<<<sbl, TB, 0, stream>>>(row, col, lw, 1.f, cbuf, t1, E);
    gemm_wmma<<<ggrid(48), 128, 0, stream>>>(t1, LD, p_c2[1], nullptr, fglob, LD, Mtiles, 64, 48, FLAG_ACCUM);
    hipMemsetAsync(t2, 0, NLD_bytes, stream);
    scatter_edges<<<sbl, TB, 0, stream>>>(row, col, lw, 2.f, t1, t2, E);
    sub_kernel<<<cdiv(N * LD, TB), TB, 0, stream>>>(t2, cbuf, N * LD);
    gemm_wmma<<<ggrid(48), 128, 0, stream>>>(t2, LD, p_c2[2], cheb2_b, fglob, LD, Mtiles, 64, 48,
                                             FLAG_ACCUM | FLAG_BIAS | FLAG_RELU);

    // ---- fusion ----
    concat_kernel<<<cdiv(N * 96, TB), TB, 0, stream>>>(floc, fglob, fcat, N);
    gemm_wmma<<<ggrid(48), 128, 0, stream>>>(fcat, 96, p_fu, fuse_b, fused, LD, Mtiles, 96, 48,
                                             FLAG_BIAS | FLAG_RELU);

    // ---- mean pool + classifier ----
    pool_accum<<<nbl, TB, 0, stream>>>(fused, batch, psum, pcnt, N);
    pool_div<<<cdiv(G * 48, TB), TB, 0, stream>>>(psum, pcnt, pooled, G);
    cls_head<<<cdiv(G * NCLS, TB), TB, 0, stream>>>(pooled, cls_w0, cls_b0, cls_w1, cls_b1,
                                                    logits, G, 48, NCLS);
}
